// GCRNN_35914516529712
// MI455X (gfx1250) — matrix operand order repaired and checked by reading, not compiled, not run on hardware
//
#include <hip/hip_runtime.h>
#include <hip/hip_bf16.h>
#include <cmath>

// ---------------- problem constants ----------------
constexpr int B_   = 128;
constexpr int T_   = 12;
constexpr int N_   = 1024;   // nodes
constexpr int U_   = 64;     // units
constexpr int DIN  = 2;
constexpr int DCAT = DIN + U_;     // 66
constexpr int DP   = 96;           // padded feature dim (3 K-tiles of 32)
constexpr int NCOL = B_ * DP;      // 12288 GEMM columns for diffusion
constexpr int RROWS = N_ * B_;     // 131072 rows for projection GEMMs

typedef __attribute__((ext_vector_type(16))) _Float16 v16h;
typedef __attribute__((ext_vector_type(8)))  _Float16 v8h;
typedef __attribute__((ext_vector_type(8)))  float    v8f;
typedef __attribute__((ext_vector_type(4)))  unsigned int v4u;
typedef __attribute__((ext_vector_type(8)))  int v8i;
typedef __attribute__((ext_vector_type(4)))  int v4i;

#if __has_builtin(__builtin_amdgcn_tensor_load_to_lds)
#define HAVE_TDM 1
#else
#define HAVE_TDM 0
#endif

// ---------------- WMMA fragment helpers (wave32, gfx1250) ----------------
static __device__ inline v16h load_frag_a(const _Float16* base, int lda, int lane, int k0) {
    int m  = lane & 15;
    int hf = lane >> 4;
    const _Float16* p = base + (size_t)m * lda + k0 + 8 * hf;
    v8h lo = *(const v8h*)(p);
    v8h hi = *(const v8h*)(p + 16);
    v16h r;
#pragma unroll
    for (int i = 0; i < 8; ++i) { r[i] = lo[i]; r[i + 8] = hi[i]; }
    return r;
}

static __device__ inline v16h load_frag_b(const _Float16* base, int ldb, int lane, int k0) {
    int n  = lane & 15;
    int kh = lane >> 4;
    const _Float16* p = base + (size_t)n * ldb + k0 + 8 * kh;
    v8h lo = *(const v8h*)(p);
    v8h hi = *(const v8h*)(p + 16);
    v16h r;
#pragma unroll
    for (int i = 0; i < 8; ++i) { r[i] = lo[i]; r[i + 8] = hi[i]; }
    return r;
}

static __device__ inline v8f wmma16(v16h a, v16h b, v8f c) {
    return __builtin_amdgcn_wmma_f32_16x16x32_f16(false, a, false, b, (short)0, c, false, false);
}

// ---------------- TDM 2D tile load: global -> LDS ----------------
static __device__ inline void tdm_load_tile(void* lds, const void* gsrc,
                                            unsigned tile_w, unsigned tile_h,
                                            unsigned long long row_stride_elems,
                                            unsigned tdim0, unsigned tdim1) {
#if HAVE_TDM
    unsigned long long ga = (unsigned long long)(size_t)gsrc;
    v4u g0 = {};
    g0[0] = 1u;                                         // count=1, user descriptor
    g0[1] = (unsigned)(size_t)lds;                      // lds_addr (bytes)
    g0[2] = (unsigned)ga;                               // global_addr[31:0]
    g0[3] = (unsigned)((ga >> 32) & 0x01FFFFFFu) | (2u << 30);  // addr[56:32] | type=2
    v8i g1 = {};
    g1[0] = (int)(1u << 16);                            // data_size = 2 bytes
    g1[1] = (int)((tdim0 & 0xFFFFu) << 16);             // tensor_dim0[15:0] @ bits 63:48
    g1[2] = (int)(((tdim0 >> 16) & 0xFFFFu) | ((tdim1 & 0xFFFFu) << 16));
    g1[3] = (int)(((tdim1 >> 16) & 0xFFFFu) | (tile_w << 16));   // tile_dim0 @127:112
    g1[4] = (int)(tile_h & 0xFFFFu);                    // tile_dim1 ; tile_dim2=0
    g1[5] = (int)(unsigned)(row_stride_elems & 0xFFFFFFFFull);   // dim0_stride lo
    g1[6] = (int)(unsigned)((row_stride_elems >> 32) & 0xFFFFull);
    g1[7] = 0;
    v4i z4 = {};
#if __clang_major__ >= 23
    v8i z8 = {};
    __builtin_amdgcn_tensor_load_to_lds(g0, g1, z4, z4, z8, 0);
#else
    __builtin_amdgcn_tensor_load_to_lds(g0, g1, z4, z4, 0);
#endif
#else
    (void)lds; (void)gsrc; (void)tile_w; (void)tile_h;
    (void)row_stride_elems; (void)tdim0; (void)tdim1;
#endif
}

static __device__ inline void tdm_wait0() {
#if HAVE_TDM
#if __has_builtin(__builtin_amdgcn_s_wait_tensorcnt)
    __builtin_amdgcn_s_wait_tensorcnt(0);
#else
    asm volatile("s_wait_tensorcnt 0x0" ::: "memory");
#endif
#endif
}

// ---------------- small prep kernels ----------------
__global__ void k_zero_f32(float* p, int n) {
    int i = blockIdx.x * blockDim.x + threadIdx.x;
    if (i < n) p[i] = 0.0f;
}

__global__ void k_cvt_s(const float* __restrict__ S, _Float16* __restrict__ S16) {
    int i = blockIdx.x * blockDim.x + threadIdx.x;   // N_*N_ threads
    S16[i] = (_Float16)S[i];
}

__global__ void k_cvt_wg(const float* __restrict__ Wg, _Float16* __restrict__ WgT) {
    int i = blockIdx.x * blockDim.x + threadIdx.x;   // 128*96
    if (i >= 128 * DP) return;
    int g = i / DP, k = i % DP;
    WgT[i] = (k < DCAT) ? (_Float16)Wg[k * 128 + g] : (_Float16)0.0f;
}

__global__ void k_cvt_wc(const float* __restrict__ Wc, _Float16* __restrict__ WcT) {
    int i = blockIdx.x * blockDim.x + threadIdx.x;   // 64*96
    if (i >= 64 * DP) return;
    int c = i / DP, k = i % DP;
    WcT[i] = (k < DCAT) ? (_Float16)Wc[k * 64 + c] : (_Float16)0.0f;
}

// XcatT[j][m], j = b*DP + dp : cat = [x_t | h]
__global__ void k_pack1(const float* __restrict__ x, const float* __restrict__ h,
                        _Float16* __restrict__ Xc, int t) {
    int idx = blockIdx.x * blockDim.x + threadIdx.x;   // NCOL*N_
    int m = idx & (N_ - 1);
    int j = idx >> 10;
    int b = j / DP, dp = j % DP;
    float v = 0.0f;
    if (dp < DIN)       v = x[(((size_t)b * T_ + t) * N_ + m) * DIN + dp];
    else if (dp < DCAT) v = h[(((size_t)b << 10) + m) * U_ + (dp - DIN)];
    Xc[idx] = (_Float16)v;
}

// cat_c = [x_t | r .* h]
__global__ void k_pack2(const float* __restrict__ x, const float* __restrict__ h,
                        const _Float16* __restrict__ gates, _Float16* __restrict__ Xc, int t) {
    int idx = blockIdx.x * blockDim.x + threadIdx.x;   // NCOL*N_
    int m = idx & (N_ - 1);
    int j = idx >> 10;
    int b = j / DP, dp = j % DP;
    float v = 0.0f;
    if (dp < DIN) {
        v = x[(((size_t)b * T_ + t) * N_ + m) * DIN + dp];
    } else if (dp < DCAT) {
        int u = dp - DIN;
        float rg = (float)gates[((size_t)(m * B_ + b)) * 128 + u];   // r gate
        v = rg * h[(((size_t)b << 10) + m) * U_ + u];
    }
    Xc[idx] = (_Float16)v;
}

// ---------------- diffusion GEMM: conv[n][j] = sum_m S16[n][m] * XcatT[j][m] ----------------
// 4-wave block computes a 128x64 tile (wave = 32x64). A (128x32) and B (64x32)
// K-slices staged in LDS by TDM (double buffered; waves 0/1 issue A/B DMAs in
// parallel). All fragments hoisted before the 8-WMMA chain -> one DS wait per
// K-step, WMMAs issue back-to-back.
__global__ void __launch_bounds__(128) k_gemm_diff(const _Float16* __restrict__ S16,
                                                   const _Float16* __restrict__ Xc,
                                                   _Float16* __restrict__ conv) {
    __shared__ _Float16 lA[2][128][32];   // 16 KB
    __shared__ _Float16 lB[2][64][32];    //  8 KB

    const int tid  = threadIdx.x;
    const int lane = tid & 31;
    const int w    = tid >> 5;
    constexpr int JB = NCOL / 64;            // 192 column-tiles
    const int mbase = (int)(blockIdx.x / JB) * 128;
    const int jbase = (int)(blockIdx.x % JB) * 64;

    const _Float16* gA = S16 + (size_t)mbase * N_;
    const _Float16* gB = Xc + (size_t)jbase * N_;

#if !HAVE_TDM
    const int brow = tid >> 1;               // fallback copy mapping
    const int bseg = (tid & 1) * 16;
#endif

    // ---- stage tile 0 into buffer 0 ----
#if HAVE_TDM
    if (w == 0)      tdm_load_tile(&lA[0][0][0], gA, 32, 128, N_, N_, 1u << 20);
    else if (w == 1) tdm_load_tile(&lB[0][0][0], gB, 32, 64, N_, N_, 1u << 20);
    if (w < 2) tdm_wait0();
#else
    *(v8h*)&lA[0][tid][0]          = *(const v8h*)(gA + (size_t)tid * N_);
    *(v8h*)&lA[0][tid][16]         = *(const v8h*)(gA + (size_t)tid * N_ + 16);
    *(v8h*)&lB[0][brow][bseg]      = *(const v8h*)(gB + (size_t)brow * N_ + bseg);
#endif
    __syncthreads();

    v8f acc[2][4] = {};
    constexpr int KT = N_ / 32;              // 32 K-steps
#pragma unroll 1
    for (int kt = 0; kt < KT; ++kt) {
        const int cur = kt & 1;
        const int nxt = cur ^ 1;
        if (kt + 1 < KT) {
            const int k0n = (kt + 1) * 32;
#if HAVE_TDM
            if (w == 0)      tdm_load_tile(&lA[nxt][0][0], gA + k0n, 32, 128, N_, N_, 1u << 20);
            else if (w == 1) tdm_load_tile(&lB[nxt][0][0], gB + k0n, 32, 64, N_, N_, 1u << 20);
#else
            *(v8h*)&lA[nxt][tid][0]     = *(const v8h*)(gA + (size_t)tid * N_ + k0n);
            *(v8h*)&lA[nxt][tid][16]    = *(const v8h*)(gA + (size_t)tid * N_ + k0n + 16);
            *(v8h*)&lB[nxt][brow][bseg] = *(const v8h*)(gB + (size_t)brow * N_ + k0n + bseg);
#endif
        }
        // hoist all fragments, then 8 WMMAs back-to-back
        v16h a0 = load_frag_a(&lA[cur][w * 32][0], 32, lane, 0);
        v16h a1 = load_frag_a(&lA[cur][w * 32 + 16][0], 32, lane, 0);
        v16h b0 = load_frag_b(&lB[cur][0][0], 32, lane, 0);
        v16h b1 = load_frag_b(&lB[cur][16][0], 32, lane, 0);
        v16h b2 = load_frag_b(&lB[cur][32][0], 32, lane, 0);
        v16h b3 = load_frag_b(&lB[cur][48][0], 32, lane, 0);
        acc[0][0] = wmma16(a0, b0, acc[0][0]);
        acc[0][1] = wmma16(a0, b1, acc[0][1]);
        acc[0][2] = wmma16(a0, b2, acc[0][2]);
        acc[0][3] = wmma16(a0, b3, acc[0][3]);
        acc[1][0] = wmma16(a1, b0, acc[1][0]);
        acc[1][1] = wmma16(a1, b1, acc[1][1]);
        acc[1][2] = wmma16(a1, b2, acc[1][2]);
        acc[1][3] = wmma16(a1, b3, acc[1][3]);
#if HAVE_TDM
        if (w < 2) tdm_wait0();
#endif
        __syncthreads();
    }

    const int hf = lane >> 4, nn = lane & 15;
#pragma unroll
    for (int rt = 0; rt < 2; ++rt)
#pragma unroll
        for (int tt = 0; tt < 4; ++tt)
#pragma unroll
            for (int r = 0; r < 8; ++r)
                conv[(size_t)(mbase + w * 32 + rt * 16 + r + 8 * hf) * NCOL +
                     jbase + tt * 16 + nn] = (_Float16)acc[rt][tt][r];
}

// ---------------- gates GEMM + sigmoid: rows (n,b), K=96, N=128 ----------------
__global__ void __launch_bounds__(128) k_gemm_gates(const _Float16* __restrict__ conv,
                                                    const _Float16* __restrict__ WgT,
                                                    const float* __restrict__ bg,
                                                    _Float16* __restrict__ gates) {
    int wave = blockIdx.x * (blockDim.x >> 5) + (threadIdx.x >> 5);
    int lane = threadIdx.x & 31;
    int rbase = (wave >> 1) * 16;
    int gbase = (wave & 1) * 64;

    const _Float16* Ab = conv + (size_t)rbase * DP;
    const _Float16* Bb = WgT + (size_t)gbase * DP;

    v8f acc[4] = {};
#pragma unroll
    for (int k0 = 0; k0 < DP; k0 += 32) {
        v16h a = load_frag_a(Ab, DP, lane, k0);
        v16h b0 = load_frag_b(Bb + (size_t)0 * DP, DP, lane, k0);
        v16h b1 = load_frag_b(Bb + (size_t)16 * DP, DP, lane, k0);
        v16h b2 = load_frag_b(Bb + (size_t)32 * DP, DP, lane, k0);
        v16h b3 = load_frag_b(Bb + (size_t)48 * DP, DP, lane, k0);
        acc[0] = wmma16(a, b0, acc[0]);
        acc[1] = wmma16(a, b1, acc[1]);
        acc[2] = wmma16(a, b2, acc[2]);
        acc[3] = wmma16(a, b3, acc[3]);
    }
    int hf = lane >> 4, nn = lane & 15;
#pragma unroll
    for (int tt = 0; tt < 4; ++tt)
#pragma unroll
        for (int r = 0; r < 8; ++r) {
            int g = gbase + tt * 16 + nn;
            int row = rbase + r + 8 * hf;
            float v = acc[tt][r] + bg[g];
            float s = 1.0f / (1.0f + __expf(-v));
            gates[(size_t)row * 128 + g] = (_Float16)s;
        }
}

// ---------------- candidate GEMM + tanh + GRU update ----------------
__global__ void __launch_bounds__(128) k_gemm_cand(const _Float16* __restrict__ conv,
                                                   const _Float16* __restrict__ WcT,
                                                   const float* __restrict__ bc,
                                                   const _Float16* __restrict__ gates,
                                                   float* __restrict__ h,
                                                   float* __restrict__ out, int t) {
    int wave = blockIdx.x * (blockDim.x >> 5) + (threadIdx.x >> 5);
    int lane = threadIdx.x & 31;
    int rbase = wave * 16;

    const _Float16* Ab = conv + (size_t)rbase * DP;

    v8f acc[4] = {};
#pragma unroll
    for (int k0 = 0; k0 < DP; k0 += 32) {
        v16h a = load_frag_a(Ab, DP, lane, k0);
        v16h b0 = load_frag_b(WcT + (size_t)0 * DP, DP, lane, k0);
        v16h b1 = load_frag_b(WcT + (size_t)16 * DP, DP, lane, k0);
        v16h b2 = load_frag_b(WcT + (size_t)32 * DP, DP, lane, k0);
        v16h b3 = load_frag_b(WcT + (size_t)48 * DP, DP, lane, k0);
        acc[0] = wmma16(a, b0, acc[0]);
        acc[1] = wmma16(a, b1, acc[1]);
        acc[2] = wmma16(a, b2, acc[2]);
        acc[3] = wmma16(a, b3, acc[3]);
    }
    int hf = lane >> 4, nn = lane & 15;
#pragma unroll
    for (int tt = 0; tt < 4; ++tt)
#pragma unroll
        for (int r = 0; r < 8; ++r) {
            int c = tt * 16 + nn;
            int row = rbase + r + 8 * hf;      // row = n*128 + b
            int n = row >> 7, b = row & (B_ - 1);
            float cv = tanhf(acc[tt][r] + bc[c]);
            float u  = (float)gates[(size_t)row * 128 + 64 + c];
            size_t hidx = (((size_t)b << 10) + n) * U_ + c;
            float hn = u * h[hidx] + (1.0f - u) * cv;
            h[hidx] = hn;
            out[(((size_t)b * T_ + t) * N_ + n) * U_ + c] = hn;
        }
}

// ---------------- host ----------------
extern "C" void kernel_launch(void* const* d_in, const int* in_sizes, int n_in,
                              void* d_out, int out_size, void* d_ws, size_t ws_size,
                              hipStream_t stream) {
    const float* x  = (const float*)d_in[0];
    const float* S  = (const float*)d_in[1];
    const float* Wg = (const float*)d_in[2];
    const float* bg = (const float*)d_in[3];
    const float* Wc = (const float*)d_in[4];
    const float* bc = (const float*)d_in[5];
    float* out = (float*)d_out;

    char* ws = (char*)d_ws;
    constexpr size_t off_S16   = 0;
    constexpr size_t off_WgT   = off_S16 + (size_t)N_ * N_ * 2;           // 2 MB
    constexpr size_t off_WcT   = off_WgT + (size_t)128 * DP * 2;
    constexpr size_t off_Xc    = off_WcT + (size_t)64 * DP * 2;
    constexpr size_t off_conv  = off_Xc + (size_t)NCOL * N_ * 2;          // 24 MB
    constexpr size_t off_gates = off_conv + (size_t)N_ * NCOL * 2;        // 24 MB
    constexpr size_t off_h     = off_gates + (size_t)RROWS * 128 * 2;     // 32 MB

    _Float16* S16   = (_Float16*)(ws + off_S16);
    _Float16* WgT   = (_Float16*)(ws + off_WgT);
    _Float16* WcT   = (_Float16*)(ws + off_WcT);
    _Float16* Xc    = (_Float16*)(ws + off_Xc);
    _Float16* conv  = (_Float16*)(ws + off_conv);
    _Float16* gates = (_Float16*)(ws + off_gates);
    float*    hbuf  = (float*)(ws + off_h);

    const int HN = B_ * N_ * U_;                       // 8388608
    k_zero_f32<<<(HN + 255) / 256, 256, 0, stream>>>(hbuf, HN);
    k_cvt_s<<<(N_ * N_) / 256, 256, 0, stream>>>(S, S16);
    k_cvt_wg<<<(128 * DP + 255) / 256, 256, 0, stream>>>(Wg, WgT);
    k_cvt_wc<<<(64 * DP + 255) / 256, 256, 0, stream>>>(Wc, WcT);

    const int packGrid = (NCOL * N_) / 256;            // 49152
    const int diffGrid = (N_ / 128) * (NCOL / 64);     // 1536 blocks x 128 thr
    const int gateGrid = ((RROWS / 16) * 2) / 4;       // 4096
    const int candGrid = (RROWS / 16) / 4;             // 2048

    for (int t = 0; t < T_; ++t) {
        k_pack1<<<packGrid, 256, 0, stream>>>(x, hbuf, Xc, t);
        k_gemm_diff<<<diffGrid, 128, 0, stream>>>(S16, Xc, conv);
        k_gemm_gates<<<gateGrid, 128, 0, stream>>>(conv, WgT, bg, gates);
        k_pack2<<<packGrid, 256, 0, stream>>>(x, hbuf, gates, Xc, t);
        k_gemm_diff<<<diffGrid, 128, 0, stream>>>(S16, Xc, conv);
        k_gemm_cand<<<candGrid, 128, 0, stream>>>(conv, WcT, bc, gates, hbuf, out, t);
    }
}
